// MultiBoxLoss_60275571032179
// MI455X (gfx1250) — compile-verified, matrix operand order checked by reading, not run on hardware
//
#include <hip/hip_runtime.h>
#include <math.h>

typedef __attribute__((ext_vector_type(16))) _Float16 v16h;
typedef __attribute__((ext_vector_type(8)))  float    v8f;
typedef int asv4i __attribute__((__vector_size__(16)));   // matches builtin param

#define BB   64
#define OO   12
#define PP   8732
#define CC   81
#define RW   85            // row dwords: 4 loc + 81 classes
#define BP   (BB * PP)     // 558848, divisible by 16
#define NG   (BP / 16)     // 34928 wave-groups, divisible by 8
#define GRPD (16 * RW)     // 1360 dwords staged per wave

#if __has_builtin(__builtin_amdgcn_global_load_async_to_lds_b128)
#define USE_ASYNC_LDS 1
#else
#define USE_ASYNC_LDS 0
#endif

// ---------------------------------------------------------------------------
// Kernel 0: zero accumulators
// ---------------------------------------------------------------------------
__global__ void k_init(int* __restrict__ posc, float* __restrict__ acc) {
  int i = threadIdx.x;
  if (i < BB) posc[i] = 0;
  if (i < 4)  acc[i]  = 0.0f;
}

// ---------------------------------------------------------------------------
// Kernel 1: best_prior_idx[b][o] = first argmax over P of IoU(truth, prior_pf)
// ---------------------------------------------------------------------------
__global__ void k_bestprior(const float* __restrict__ truths,
                            const float* __restrict__ dbox,
                            int* __restrict__ bp) {
  __shared__ float sv[128];
  __shared__ int   si[128];
  const int b = blockIdx.x / OO, o = blockIdx.x % OO;
  const float4 t = ((const float4*)truths)[b * OO + o];
  const float at = (t.z - t.x) * (t.w - t.y);
  float best = -1.0f; int bidx = 0;
  for (int p = threadIdx.x; p < PP; p += blockDim.x) {
    float4 d  = ((const float4*)dbox)[p];
    float px1 = d.x - 0.5f * d.z, py1 = d.y - 0.5f * d.w;
    float px2 = d.x + 0.5f * d.z, py2 = d.y + 0.5f * d.w;
    float ix  = fmaxf(fminf(t.z, px2) - fmaxf(t.x, px1), 0.0f);
    float iy  = fmaxf(fminf(t.w, py2) - fmaxf(t.y, py1), 0.0f);
    float inter = ix * iy;
    float ap  = (px2 - px1) * (py2 - py1);
    float iou = inter / (at + ap - inter);
    if (iou > best) { best = iou; bidx = p; }   // ascending p -> first max
  }
  sv[threadIdx.x] = best; si[threadIdx.x] = bidx;
  __syncthreads();
  for (int s = blockDim.x / 2; s > 0; s >>= 1) {
    if (threadIdx.x < s) {
      float v2 = sv[threadIdx.x + s]; int i2 = si[threadIdx.x + s];
      if (v2 > sv[threadIdx.x] ||
          (v2 == sv[threadIdx.x] && i2 < si[threadIdx.x])) {
        sv[threadIdx.x] = v2; si[threadIdx.x] = i2;
      }
    }
    __syncthreads();
  }
  if (threadIdx.x == 0) bp[b * OO + o] = si[0];
}

// ---------------------------------------------------------------------------
// Kernel 2: fused match + smooth-L1 + CE.  One wave handles 16 (b,p) rows:
//   - async (or coalesced float4) staging of 16 prediction rows into LDS
//   - per-lane match/encode (lane pair L, L+16 mirrors prior L%16)
//   - exp-sum over 81 classes via V_WMMA_F32_16X16X32_F16 with B = ones
// ---------------------------------------------------------------------------
__global__ void __launch_bounds__(256, 2)
k_main(const float* __restrict__ pred,
       const float* __restrict__ truths,
       const int*   __restrict__ labels,
       const float* __restrict__ dbox,
       const int*   __restrict__ bp,
       float* __restrict__ ceneg,
       int*   __restrict__ posc,
       float* __restrict__ acc) {
  __shared__ float smem[8 * GRPD];
  const int lane = threadIdx.x & 31;
  const int wv   = threadIdx.x >> 5;
  const int g    = blockIdx.x * 8 + wv;
  if (g >= NG) return;                       // wave-uniform

  float* lbuf = smem + wv * GRPD;

  // ---- stage 16 rows (5440 B, 16B-aligned since base = 5440*g) ----
#if USE_ASYNC_LDS
  {
    const float4* src = (const float4*)pred + (size_t)g * (GRPD / 4);
    float4* dst = (float4*)lbuf;
#pragma unroll
    for (int t = 0; t < GRPD / 4 / 32; ++t) {     // 10 full rounds
      int idx = lane + 32 * t;
      __builtin_amdgcn_global_load_async_to_lds_b128(
          (asv4i*)(src + idx), (asv4i*)(dst + idx), 0, 0);
    }
    int rem = lane + 32 * (GRPD / 4 / 32);        // 20-lane tail
    if (rem < GRPD / 4)
      __builtin_amdgcn_global_load_async_to_lds_b128(
          (asv4i*)(src + rem), (asv4i*)(dst + rem), 0, 0);
  }
  __builtin_amdgcn_wave_barrier();
#if __has_builtin(__builtin_amdgcn_s_wait_asynccnt)
  __builtin_amdgcn_s_wait_asynccnt(0);
#else
  asm volatile("s_wait_asynccnt 0" ::: "memory");
#endif
  asm volatile("" ::: "memory");
#else
  {
    const float4* src = (const float4*)pred + (size_t)g * (GRPD / 4);
    float4* dst = (float4*)lbuf;
#pragma unroll
    for (int t = 0; t < GRPD / 4 / 32; ++t)   // 10 full rounds
      dst[lane + 32 * t] = src[lane + 32 * t];
    int rem = lane + 32 * (GRPD / 4 / 32);    // 20-lane tail
    if (rem < GRPD / 4) dst[rem] = src[rem];
  }
  __builtin_amdgcn_wave_barrier();
  asm volatile("s_wait_dscnt 0" ::: "memory");   // cross-lane LDS visibility
#endif

  const int q = lane & 15;                   // prior slot within group
  int flat = g * 16 + q;
  const bool active = flat < BP;             // always true (exact division)
  if (flat >= BP) flat = BP - 1;
  const int b = flat / PP;
  const int p = flat - b * PP;

  // prior (cx,cy,w,h) and its point form
  const float4 d = ((const float4*)dbox)[p];
  const float px1 = d.x - 0.5f * d.z, py1 = d.y - 0.5f * d.w;
  const float px2 = d.x + 0.5f * d.z, py2 = d.y + 0.5f * d.w;
  const float ap  = (px2 - px1) * (py2 - py1);

  // best truth for this prior (first argmax over o)
  float bov = -1.0f; int bt = 0;
  const float4* tb = (const float4*)truths + b * OO;
#pragma unroll
  for (int o = 0; o < OO; ++o) {
    float4 t = tb[o];
    float ix = fmaxf(fminf(t.z, px2) - fmaxf(t.x, px1), 0.0f);
    float iy = fmaxf(fminf(t.w, py2) - fmaxf(t.y, py1), 0.0f);
    float inter = ix * iy;
    float at = (t.z - t.x) * (t.w - t.y);
    float iou = inter / (at + ap - inter);
    if (iou > bov) { bov = iou; bt = o; }
  }
  // forced matches (ascending o: last write wins)
#pragma unroll
  for (int o = 0; o < OO; ++o)
    if (bp[b * OO + o] == p) { bt = o; bov = 2.0f; }

  int tgt = 0;
  if (bov >= 0.5f) tgt = labels[b * OO + bt] + 1;
  const bool pos = tgt > 0;

  const float* lrow = lbuf + q * RW;

  // target logit: one dynamically-indexed LDS dword
  const float xt = lrow[4 + tgt];

  // smooth-L1 localization loss (positives only)
  float sl1 = 0.0f;
  if (pos) {
    float4 m = tb[bt];
    float lt0 = (0.5f * (m.x + m.z) - d.x) / (0.1f * d.z);
    float lt1 = (0.5f * (m.y + m.w) - d.y) / (0.1f * d.w);
    float lt2 = __logf((m.z - m.x) / d.z) * 5.0f;    // / 0.2
    float lt3 = __logf((m.w - m.y) / d.w) * 5.0f;
    float lt[4] = {lt0, lt1, lt2, lt3};
#pragma unroll
    for (int i = 0; i < 4; ++i) {
      float df = lrow[i] - lt[i];
      float adf = fabsf(df);
      sl1 += (adf < 1.0f) ? 0.5f * df * df : adf - 0.5f;
    }
  }

  // ---- logsumexp over 81 classes: WMMA with B = ones ----
  v16h bones;
#pragma unroll
  for (int e = 0; e < 16; ++e) bones[e] = (_Float16)1.0f;
  v8f csum = {0.f, 0.f, 0.f, 0.f, 0.f, 0.f, 0.f, 0.f};
  const int kbase = (lane < 16) ? 0 : 8;     // ISA 16-bit A layout
#pragma unroll
  for (int ch = 0; ch < 3; ++ch) {
    v16h a;
#pragma unroll
    for (int e = 0; e < 16; ++e) {
      int k = ch * 32 + kbase + ((e < 8) ? e : (e + 8));
      a[e] = (k < CC) ? (_Float16)__expf(lrow[4 + k]) : (_Float16)0.0f;
    }
    csum = __builtin_amdgcn_wmma_f32_16x16x32_f16(
        false, a, false, bones, (short)0, csum, false, false);
  }

  // D row for prior q: M<8 lives in lanes 0..15, M>=8 in lanes 16..31;
  // every column equals the sum, so any column works.
  const int r = q & 7;
  float so = csum[0];
  if (r == 1) so = csum[1];
  if (r == 2) so = csum[2];
  if (r == 3) so = csum[3];
  if (r == 4) so = csum[4];
  if (r == 5) so = csum[5];
  if (r == 6) so = csum[6];
  if (r == 7) so = csum[7];
  float sx = __shfl_xor(so, 16, 32);
  bool own = ((lane & 8) != 0) == ((lane & 16) != 0);
  float ssum = own ? so : sx;
  float ce = __logf(ssum) - xt;

  if (active && lane < 16) {
    ceneg[(size_t)b * PP + p] = pos ? 0.0f : ce;
    if (pos) {
      atomicAdd(&posc[b], 1);
      atomicAdd(&acc[0], sl1);               // loss_l
      atomicAdd(&acc[1], ce);                // positive CE part of loss_c
    }
  }
}

// ---------------------------------------------------------------------------
// Kernel 3: exact top-k sum per batch via 32-bit radix select (all ce >= 0)
// ---------------------------------------------------------------------------
__global__ void k_topk(const float* __restrict__ ceneg,
                       const int* __restrict__ posc,
                       float* __restrict__ acc) {
  __shared__ unsigned cnt[16];
  __shared__ unsigned sh_prefix, sh_krem;
  __shared__ float red[256];
  const int b = blockIdx.x;
  const float* row = ceneg + (size_t)b * PP;
  int k = posc[b] * 3;
  if (k > PP) k = PP;
  if (k <= 0) return;
  if (threadIdx.x == 0) { sh_prefix = 0u; sh_krem = (unsigned)k; }
  __syncthreads();
  for (int nb = 28; nb >= 0; nb -= 4) {
    if (threadIdx.x < 16) cnt[threadIdx.x] = 0u;
    __syncthreads();
    unsigned prefix = sh_prefix;
    unsigned hm = (nb + 4 >= 32) ? 0u : (0xFFFFFFFFu << (nb + 4));
    for (int i = threadIdx.x; i < PP; i += blockDim.x) {
      unsigned u = __float_as_uint(row[i]);
      if ((u & hm) == prefix) atomicAdd(&cnt[(u >> nb) & 15], 1u);
    }
    __syncthreads();
    if (threadIdx.x == 0) {
      unsigned krem = sh_krem;
      for (int dgt = 15; dgt >= 0; --dgt) {
        if (cnt[dgt] >= krem || dgt == 0) {
          sh_prefix = prefix | ((unsigned)dgt << nb);
          break;
        }
        krem -= cnt[dgt];
      }
      sh_krem = krem;
    }
    __syncthreads();
  }
  const unsigned T = sh_prefix;
  const unsigned krem = sh_krem;
  const float thr = __uint_as_float(T);
  float s = 0.0f;
  for (int i = threadIdx.x; i < PP; i += blockDim.x) {
    float v = row[i];
    if (__float_as_uint(v) > T) s += v;
  }
  red[threadIdx.x] = s;
  __syncthreads();
  for (int st = blockDim.x / 2; st > 0; st >>= 1) {
    if (threadIdx.x < st) red[threadIdx.x] += red[threadIdx.x + st];
    __syncthreads();
  }
  if (threadIdx.x == 0)
    atomicAdd(&acc[2], red[0] + (float)krem * thr);   // top-k negatives sum
}

// ---------------------------------------------------------------------------
// Kernel 4: finalize the two scalars
// ---------------------------------------------------------------------------
__global__ void k_final(const int* __restrict__ posc,
                        const float* __restrict__ acc,
                        float* __restrict__ out) {
  if (threadIdx.x == 0 && blockIdx.x == 0) {
    int n = 0;
    for (int b = 0; b < BB; ++b) n += posc[b];
    float N = (float)n;
    out[0] = acc[0] / N;
    out[1] = (acc[1] + acc[2]) / N;
  }
}

// ---------------------------------------------------------------------------
extern "C" void kernel_launch(void* const* d_in, const int* in_sizes, int n_in,
                              void* d_out, int out_size, void* d_ws,
                              size_t ws_size, hipStream_t stream) {
  const float* pred   = (const float*)d_in[0];   // (64, 8732, 85) f32
  const float* truths = (const float*)d_in[1];   // (64, 12, 4)    f32
  const int*   labels = (const int*)d_in[2];     // (64, 12)       i32
  const float* dbox   = (const float*)d_in[3];   // (8732, 4)      f32
  float* out = (float*)d_out;                    // [loss_l/N, loss_c/N]

  char* ws = (char*)d_ws;
  float* ceneg = (float*)ws;                         // BP floats (~2.24 MB)
  int*   bp    = (int*)(ws + sizeof(float) * (size_t)BP);
  int*   posc  = bp + BB * OO;
  float* acc   = (float*)(posc + BB);                // [loss_l, ce_pos, ce_negtopk]

  k_init<<<1, 128, 0, stream>>>(posc, acc);
  k_bestprior<<<BB * OO, 128, 0, stream>>>(truths, dbox, bp);
  k_main<<<NG / 8, 256, 0, stream>>>(pred, truths, labels, dbox, bp,
                                     ceneg, posc, acc);
  k_topk<<<BB, 256, 0, stream>>>(ceneg, posc, acc);
  k_final<<<1, 1, 0, stream>>>(posc, acc, out);
}
